// Highlighter_69166153334853
// MI455X (gfx1250) — matrix operand order repaired and checked by reading, not compiled, not run on hardware
//
#include <hip/hip_runtime.h>

#define NUM_WORDS 36
#define NUM_SELECT 4
#define NEGV (-1e30f)
#define BD 512
#define TD 256
#define DD 512

typedef __attribute__((ext_vector_type(16))) __bf16 v16bf;
typedef __attribute__((ext_vector_type(8)))  float  v8f;
typedef __attribute__((ext_vector_type(4)))  float  v4f;
typedef __attribute__((ext_vector_type(8)))  unsigned short v8u;

union BfOp { v16bf v; unsigned short u[16]; v8u h[2]; };

// Native conversion: backend emits packed v_cvt_pk_bf16_f32 (RNE).
__device__ __forceinline__ unsigned short f2bf(float f) {
  __bf16 b = (__bf16)f;
  return __builtin_bit_cast(unsigned short, b);
}

// A-operand (16x32, MxK), fp32 row-major source, converted to bf16 in-register.
// ISA layout: lanes 0-15 hold K o..o+7 (v0-3) and o+16..o+23 (v4-7), o=(lane>>4)*8.
__device__ __forceinline__ v16bf loadA_f32(const float* __restrict__ base, int ld, int kb, int lane) {
  int row = lane & 15, o = (lane >> 4) * 8;
  const float* p = base + (size_t)row * ld + kb + o;
  v4f x0 = *(const v4f*)(p);
  v4f x1 = *(const v4f*)(p + 4);
  v4f y0 = *(const v4f*)(p + 16);
  v4f y1 = *(const v4f*)(p + 20);
  v16bf r;
#pragma unroll
  for (int i = 0; i < 4; i++) {
    r[i]      = (__bf16)x0[i];
    r[4 + i]  = (__bf16)x1[i];
    r[8 + i]  = (__bf16)y0[i];
    r[12 + i] = (__bf16)y1[i];
  }
  return r;
}

// B-operand (32x16, KxN) from a row-major NxK bf16 source (i.e. computing X*Y^T).
// ISA layout: lanes 0-15 hold K o..o+7 and o+16..o+23 of column n=lane&15, o=(lane>>4)*8.
__device__ __forceinline__ v16bf loadB_bf16(const unsigned short* __restrict__ base, int ld, int kb, int lane) {
  int row = lane & 15, o = (lane >> 4) * 8;
  const unsigned short* p = base + (size_t)row * ld + kb + o;
  BfOp r;
  r.h[0] = *(const v8u*)(p);
  r.h[1] = *(const v8u*)(p + 16);
  return r.v;
}

__device__ __forceinline__ v8f wmma_bf16(v16bf a, v16bf b, v8f c) {
  return __builtin_amdgcn_wmma_f32_16x16x32_bf16(false, a, false, b, (short)0, c, false, false);
}

// ---------------- K0: weight prep (W2 -> bf16, W1 -> transposed bf16) ----------------
__global__ void __launch_bounds__(256) prep_weights(const float* __restrict__ W1,
                                                    const float* __restrict__ W2,
                                                    unsigned short* __restrict__ w1t,
                                                    unsigned short* __restrict__ w2b) {
  int i = blockIdx.x * 256 + threadIdx.x;            // 0..262143
  int d = i & (DD - 1), e = i >> 9;
  w2b[i] = f2bf(W2[i]);
  w1t[(size_t)d * DD + e] = f2bf(W1[(size_t)e * DD + d]);
}

// ---------------- K1: tf[r,e] = text_embed[r,:] . W2[e,:] + b2[e] ----------------
// M = B*N = 18432, Ncols = 512, K = 512. 4 waves/block, each wave: 16 rows x 128 cols.
__global__ void __launch_bounds__(128) gemm_textfeat(const float* __restrict__ X,
                                                     const unsigned short* __restrict__ W2b,
                                                     const float* __restrict__ bias,
                                                     float* __restrict__ tf) {
  int rtile = blockIdx.x;                            // 0..1151
  int w = threadIdx.x >> 5, lane = threadIdx.x & 31;
  const float* A = X + (size_t)rtile * 16 * DD;
  int ebase = w * 128;
  v8f acc[8];
#pragma unroll
  for (int j = 0; j < 8; j++) acc[j] = (v8f){0.f,0.f,0.f,0.f,0.f,0.f,0.f,0.f};
  for (int kb = 0; kb < DD; kb += 32) {
    v16bf a = loadA_f32(A, DD, kb, lane);
    v16bf bt[8];
#pragma unroll
    for (int j = 0; j < 8; j++)
      bt[j] = loadB_bf16(W2b + (size_t)(ebase + j * 16) * DD, DD, kb, lane);
#pragma unroll
    for (int j = 0; j < 8; j++)
      acc[j] = wmma_bf16(a, bt[j], acc[j]);
  }
  int mrow = (lane >> 4) << 3;
#pragma unroll
  for (int j = 0; j < 8; j++) {
    int e = ebase + j * 16 + (lane & 15);
    float bv = bias[e];
#pragma unroll
    for (int v = 0; v < 8; v++) {
      int r = rtile * 16 + v + mrow;
      tf[(size_t)r * DD + e] = acc[j][v] + bv;
    }
  }
}

// ---------------- K2: c[r] = b1 . tf[r,:] ----------------
__global__ void __launch_bounds__(256) cvec_kernel(const float* __restrict__ tf,
                                                   const float* __restrict__ b1,
                                                   float* __restrict__ cvec) {
  int w = threadIdx.x >> 5, lane = threadIdx.x & 31;
  int r = blockIdx.x * 8 + w;                        // 0..18431
  const float* row = tf + (size_t)r * DD;
  float s = 0.f;
  for (int e = lane; e < DD; e += 32) s += b1[e] * row[e];
#pragma unroll
  for (int i = 16; i >= 1; i >>= 1) s += __shfl_xor(s, i, 32);
  if (lane == 0) cvec[r] = s;
}

// ---------------- K3: V[r,d] = tf[r,:] . W1t[d,:]  -> bf16, padded 48-row/batch ----------------
__global__ void __launch_bounds__(128) gemm_vmat(const float* __restrict__ tf,
                                                 const unsigned short* __restrict__ w1t,
                                                 unsigned short* __restrict__ vb) {
  int rtile = blockIdx.x;                            // 0..1151
  int w = threadIdx.x >> 5, lane = threadIdx.x & 31;
  const float* A = tf + (size_t)rtile * 16 * DD;
  int dbase = w * 128;
  v8f acc[8];
#pragma unroll
  for (int j = 0; j < 8; j++) acc[j] = (v8f){0.f,0.f,0.f,0.f,0.f,0.f,0.f,0.f};
  for (int kb = 0; kb < DD; kb += 32) {
    v16bf a = loadA_f32(A, DD, kb, lane);
    v16bf bt[8];
#pragma unroll
    for (int j = 0; j < 8; j++)
      bt[j] = loadB_bf16(w1t + (size_t)(dbase + j * 16) * DD, DD, kb, lane);
#pragma unroll
    for (int j = 0; j < 8; j++)
      acc[j] = wmma_bf16(a, bt[j], acc[j]);
  }
  int mrow = (lane >> 4) << 3;
#pragma unroll
  for (int j = 0; j < 8; j++) {
    int d = dbase + j * 16 + (lane & 15);
#pragma unroll
    for (int v = 0; v < 8; v++) {
      unsigned r = (unsigned)(rtile * 16 + v + mrow);
      unsigned bb = r / 36u;
      unsigned n = r - bb * 36u;
      vb[(size_t)bb * 48 * DD + (size_t)n * DD + d] = f2bf(acc[j][v]);
    }
  }
}

// ---------------- K4: scores = temp_feat . V^T (+c), mask, softmax -> matrix ----------------
// grid (T/64, B); 4 waves/block; wave handles 16 T-rows x 48 (padded) cols.
__global__ void __launch_bounds__(128) scores_softmax_kernel(const float* __restrict__ temp_feat,
                                                             const unsigned short* __restrict__ vb,
                                                             const float* __restrict__ cvec,
                                                             const int* __restrict__ caption_len,
                                                             float* __restrict__ out_mat) {
  int b = blockIdx.y;
  int w = threadIdx.x >> 5, lane = threadIdx.x & 31;
  int ttile = blockIdx.x * 64 + w * 16;
  const float* A = temp_feat + ((size_t)b * TD + ttile) * DD;
  const unsigned short* Bm = vb + (size_t)b * 48 * DD;
  v8f acc0 = (v8f){0.f,0.f,0.f,0.f,0.f,0.f,0.f,0.f};
  v8f acc1 = acc0, acc2 = acc0;
  for (int kb = 0; kb < DD; kb += 32) {
    v16bf a  = loadA_f32(A, DD, kb, lane);
    v16bf b0 = loadB_bf16(Bm,            DD, kb, lane);
    v16bf b1 = loadB_bf16(Bm + 16 * DD,  DD, kb, lane);
    v16bf b2 = loadB_bf16(Bm + 32 * DD,  DD, kb, lane);
    acc0 = wmma_bf16(a, b0, acc0);
    acc1 = wmma_bf16(a, b1, acc1);
    acc2 = wmma_bf16(a, b2, acc2);
  }
  int cn = lane & 15;
  int cap = caption_len[b];
  float c0 = cvec[b * NUM_WORDS + cn];
  float c1 = cvec[b * NUM_WORDS + cn + 16];
  float c2 = (cn < 4) ? cvec[b * NUM_WORDS + cn + 32] : 0.f;
  bool m0 = cn >= cap, m1 = (cn + 16) >= cap, m2 = (cn + 32) >= cap;
  int mrow = (lane >> 4) << 3;
#pragma unroll
  for (int v = 0; v < 8; v++) {
    float s0 = m0 ? NEGV : acc0[v] + c0;
    float s1 = m1 ? NEGV : acc1[v] + c1;
    float s2 = m2 ? NEGV : acc2[v] + c2;
    float mx = fmaxf(s0, fmaxf(s1, s2));
#pragma unroll
    for (int i = 1; i < 16; i <<= 1) mx = fmaxf(mx, __shfl_xor(mx, i, 16));
    float e0 = __expf(s0 - mx), e1 = __expf(s1 - mx), e2 = __expf(s2 - mx);
    float sm = e0 + e1 + e2;
#pragma unroll
    for (int i = 1; i < 16; i <<= 1) sm += __shfl_xor(sm, i, 16);
    float inv = 1.0f / sm;
    int t = ttile + v + mrow;
    float* row = out_mat + ((size_t)b * TD + t) * NUM_WORDS;
    row[cn] = e0 * inv;
    row[cn + 16] = e1 * inv;
    if (cn < 4) row[cn + 32] = e2 * inv;
  }
}

// ---------------- K5: top-4 of mid row, gather tf rows -> selected ----------------
__global__ void __launch_bounds__(256) topk_select_kernel(const float* __restrict__ out_mat,
                                                          const float* __restrict__ tf,
                                                          float* __restrict__ out_sel) {
  int b = blockIdx.x;
  __shared__ int sidx[NUM_SELECT];
  if (threadIdx.x == 0) {
    float vals[NUM_WORDS];
    const float* mid = out_mat + ((size_t)b * TD + TD / 2) * NUM_WORDS;
    for (int n = 0; n < NUM_WORDS; n++) vals[n] = mid[n];
    for (int k = 0; k < NUM_SELECT; k++) {
      int bi = 0; float bv = -3.4e38f;
      for (int n = 0; n < NUM_WORDS; n++)
        if (vals[n] > bv) { bv = vals[n]; bi = n; }
      sidx[k] = bi;
      vals[bi] = -3.4e38f;
    }
  }
  __syncthreads();
  for (int k = 0; k < NUM_SELECT; k++) {
    int n = sidx[k];
    const float* src = tf + ((size_t)b * NUM_WORDS + n) * DD;
    float* dst = out_sel + ((size_t)b * NUM_SELECT + k) * DD;
    for (int e = threadIdx.x; e < DD; e += 256) dst[e] = src[e];
  }
}

extern "C" void kernel_launch(void* const* d_in, const int* in_sizes, int n_in,
                              void* d_out, int out_size, void* d_ws, size_t ws_size,
                              hipStream_t stream) {
  const float* temp_feat  = (const float*)d_in[0];
  const float* text_embed = (const float*)d_in[1];
  const int*   caption_len = (const int*)d_in[2];
  const float* W1 = (const float*)d_in[3];
  const float* b1 = (const float*)d_in[4];
  const float* W2 = (const float*)d_in[5];
  const float* b2 = (const float*)d_in[6];

  char* ws = (char*)d_ws;
  unsigned short* w1t = (unsigned short*)(ws);                       // 512KB
  unsigned short* w2b = (unsigned short*)(ws + (512u << 10));        // 512KB
  float*          tf  = (float*)(ws + (1024u << 10));                // 37.75MB
  unsigned short* vb  = (unsigned short*)(ws + (1024u << 10) + 37748736u); // 25.2MB
  float*        cvec  = (float*)(ws + (1024u << 10) + 37748736u + 25165824u); // 72KB

  float* out_sel = (float*)d_out;                                    // B*4*D
  float* out_mat = out_sel + (size_t)BD * NUM_SELECT * DD;           // B*T*N

  prep_weights<<<1024, 256, 0, stream>>>(W1, W2, w1t, w2b);
  gemm_textfeat<<<(BD * NUM_WORDS) / 16, 128, 0, stream>>>(text_embed, w2b, b2, tf);
  cvec_kernel<<<(BD * NUM_WORDS) / 8, 256, 0, stream>>>(tf, b1, cvec);
  gemm_vmat<<<(BD * NUM_WORDS) / 16, 128, 0, stream>>>(tf, w1t, vb);
  dim3 g4(TD / 64, BD);
  scores_softmax_kernel<<<g4, 128, 0, stream>>>(temp_feat, vb, cvec, caption_len, out_mat);
  topk_select_kernel<<<BD, 256, 0, stream>>>(out_mat, tf, out_sel);
}
